// TwoWordPSDProbe_71622874628395
// MI455X (gfx1250) — matrix operand order repaired and checked by reading, not compiled, not run on hardware
//
#include <hip/hip_runtime.h>

// B=4, S=4096, D=1024, R=64
// out = max(||p_i||^2 + ||p_j||^2 - 2 p_i.p_j, 0), p = x @ proj
//
// Stage 0: swizzle proj f32->f16 into WMMA B-operand layout (L2-resident, 128 KB)
// Stage 1: p = x@proj via v_wmma_f32_16x16x32_f16, emit p as f16 row-major + sq
// Stage 2: cross-GEMM p@p^T via WMMA + fused distance epilogue, NT stores (268 MB)

typedef __attribute__((ext_vector_type(16))) _Float16 v16h;
typedef __attribute__((ext_vector_type(8)))  _Float16 v8h;
typedef __attribute__((ext_vector_type(8)))  float    v8f;
typedef __attribute__((ext_vector_type(4)))  float    v4f;

#define NB 4
#define SS 4096
#define DD 1024
#define RR 64
#define NROWS (NB * SS)   // 16384

// ---------------------------------------------------------------------------
// Stage 0: proj [1024,64] f32 -> swz f16 in B-matrix lane layout.
// Layout: [kt(32)][nt(4)][lane(32)][v(16)] halves; lane<16: N=nt*16+lane, K=kt*32+v (v<8), kt*32+8+v (v>=8)
//         lane>=16: N=nt*16+(lane-16), K=kt*32+8+v (v<8), kt*32+16+v (v>=8)  (two 8-K runs per lane)
// ---------------------------------------------------------------------------
__global__ void __launch_bounds__(256) k_swz(const float* __restrict__ proj,
                                             _Float16* __restrict__ swz) {
  int tid  = blockIdx.x * 256 + threadIdx.x;      // 65536 total
  int v    = tid & 15;
  int lane = (tid >> 4) & 31;
  int nt   = (tid >> 9) & 3;
  int kt   = tid >> 11;
  int n = nt * 16 + (lane & 15);
  int k = kt * 32 + ((v >> 3) << 4) + ((lane >> 4) << 3) + (v & 7);
  swz[tid] = (_Float16)proj[k * RR + n];
}

// ---------------------------------------------------------------------------
// Stage 1: one wave -> 16 rows of p.  K loop: D=1024 in 32-steps.
// A (x tile, 16x32 f16): lane m=L%16,h=L/16 holds K {kc+8h..+7} and {kc+16+8h..+7}
// C layout: VGPR g, lane L -> (M = g+8h, N = (L%16)+nt*16)
// ---------------------------------------------------------------------------
__global__ void __launch_bounds__(256) k_proj(const float* __restrict__ x,
                                              const _Float16* __restrict__ swz,
                                              _Float16* __restrict__ p,
                                              float* __restrict__ sq) {
  const int lane = threadIdx.x & 31;
  const int wave = threadIdx.x >> 5;
  const int row0 = (blockIdx.x * 8 + wave) * 16;
  const int h    = lane >> 4;
  const int m    = lane & 15;
  const float* xrow = x + (size_t)(row0 + m) * DD;

  v8f c0 = {}, c1 = {}, c2 = {}, c3 = {};

  for (int kt = 0; kt < 32; ++kt) {
    const int kc = kt << 5;
    __builtin_prefetch(xrow + kc + 256, 0, 0);    // global_prefetch_b8, stream x
    const v4f* xa = (const v4f*)(xrow + kc);
    v4f a0 = xa[2 * h + 0];
    v4f a1 = xa[2 * h + 1];
    v4f a2 = xa[2 * h + 4];
    v4f a3 = xa[2 * h + 5];
    v16h A;
#pragma unroll
    for (int i = 0; i < 4; ++i) {
      A[i]      = (_Float16)a0[i];
      A[4 + i]  = (_Float16)a1[i];
      A[8 + i]  = (_Float16)a2[i];
      A[12 + i] = (_Float16)a3[i];
    }
    const _Float16* bb = swz + ((size_t)(kt * 4) * 32 + lane) * 16;
    v16h B0 = *(const v16h*)(bb + 0 * 512);
    v16h B1 = *(const v16h*)(bb + 1 * 512);
    v16h B2 = *(const v16h*)(bb + 2 * 512);
    v16h B3 = *(const v16h*)(bb + 3 * 512);
    c0 = __builtin_amdgcn_wmma_f32_16x16x32_f16(false, A, false, B0, (short)0, c0, false, false);
    c1 = __builtin_amdgcn_wmma_f32_16x16x32_f16(false, A, false, B1, (short)0, c1, false, false);
    c2 = __builtin_amdgcn_wmma_f32_16x16x32_f16(false, A, false, B2, (short)0, c2, false, false);
    c3 = __builtin_amdgcn_wmma_f32_16x16x32_f16(false, A, false, B3, (short)0, c3, false, false);
  }

  // Emit p (f16, row-major [row][64]) and sq = sum(f16(p)^2) per row.
  float sg[8];
#pragma unroll
  for (int g = 0; g < 8; ++g) {
    _Float16 h0 = (_Float16)c0[g];
    _Float16 h1 = (_Float16)c1[g];
    _Float16 h2 = (_Float16)c2[g];
    _Float16 h3 = (_Float16)c3[g];
    _Float16* pr = p + (size_t)(row0 + g + 8 * h) * RR + m;
    pr[0]  = h0;
    pr[16] = h1;
    pr[32] = h2;
    pr[48] = h3;
    float f0 = (float)h0, f1 = (float)h1, f2 = (float)h2, f3 = (float)h3;
    float s = f0 * f0 + f1 * f1 + f2 * f2 + f3 * f3;
#pragma unroll
    for (int off = 1; off < 16; off <<= 1) s += __shfl_xor(s, off, 32);
    sg[g] = s;
  }
  if (m == 0) {
#pragma unroll
    for (int g = 0; g < 8; ++g) sq[row0 + g + 8 * h] = sg[g];
  }
}

// ---------------------------------------------------------------------------
// Stage 2: per batch, 128x128 output tile per block; wave w -> 16x128 strip.
// A operand: p rows i (two 16B chunks/lane per k-step).
// B operand: p rows j (one 32B chunk/lane per k-step; p^T columns == p rows).
// ---------------------------------------------------------------------------
__global__ void __launch_bounds__(256) k_dist(const _Float16* __restrict__ p,
                                              const float* __restrict__ sq,
                                              float* __restrict__ out) {
  const int lane = threadIdx.x & 31;
  const int wave = threadIdx.x >> 5;
  const int b    = blockIdx.z;
  const int i0   = blockIdx.y * 128 + wave * 16;   // within batch
  const int j0   = blockIdx.x * 128;
  const int h    = lane >> 4;
  const int m    = lane & 15;
  const size_t rbase = (size_t)b * SS;

  // A tiles for K=0..31 and K=32..63
  const _Float16* arow = p + (rbase + i0 + m) * RR;
  v8h t0 = *(const v8h*)(arow + 16 * h);
  v8h t1 = *(const v8h*)(arow + 8 + 16 * h);
  v8h t2 = *(const v8h*)(arow + 32 + 16 * h);
  v8h t3 = *(const v8h*)(arow + 40 + 16 * h);
  v16h A0, A1;
#pragma unroll
  for (int i = 0; i < 8; ++i) {
    A0[i]     = t0[i];
    A0[8 + i] = t1[i];
    A1[i]     = t2[i];
    A1[8 + i] = t3[i];
  }

  const float* sqb = sq + rbase;
  float sqi[8];
#pragma unroll
  for (int g = 0; g < 8; ++g) sqi[g] = sqb[i0 + 8 * h + g];

  float* outrow = out + ((size_t)(b * SS + i0 + 8 * h) * SS + j0 + m);

#pragma unroll
  for (int nt = 0; nt < 8; ++nt) {
    const int j = j0 + nt * 16 + m;
    const _Float16* brow = p + (rbase + j) * RR;
    v16h B0 = *(const v16h*)(brow + 16 * h);        // K 0..31
    v16h B1 = *(const v16h*)(brow + 32 + 16 * h);   // K 32..63
    v8f c = {};
    c = __builtin_amdgcn_wmma_f32_16x16x32_f16(false, A0, false, B0, (short)0, c, false, false);
    c = __builtin_amdgcn_wmma_f32_16x16x32_f16(false, A1, false, B1, (short)0, c, false, false);
    const float sqj = sqb[j];
#pragma unroll
    for (int g = 0; g < 8; ++g) {
      float d = fmaxf(sqi[g] + sqj - 2.0f * c[g], 0.0f);
      __builtin_nontemporal_store(d, &outrow[(size_t)g * SS + nt * 16]);
    }
  }
}

extern "C" void kernel_launch(void* const* d_in, const int* in_sizes, int n_in,
                              void* d_out, int out_size, void* d_ws, size_t ws_size,
                              hipStream_t stream) {
  (void)in_sizes; (void)n_in; (void)out_size; (void)ws_size;
  const float* x    = (const float*)d_in[0];
  const float* proj = (const float*)d_in[1];
  float* out        = (float*)d_out;

  char* ws = (char*)d_ws;
  _Float16* pbuf = (_Float16*)ws;                                   // 16384*64*2 = 2 MiB
  float*    sqb  = (float*)(ws + (size_t)NROWS * RR * 2);           // 64 KiB
  _Float16* swz  = (_Float16*)(ws + (size_t)NROWS * RR * 2 + (size_t)NROWS * 4); // 128 KiB

  k_swz <<<256, 256, 0, stream>>>(proj, swz);
  k_proj<<<128, 256, 0, stream>>>(x, swz, pbuf, sqb);
  k_dist<<<dim3(32, 32, NB), 256, 0, stream>>>(pbuf, sqb, out);
}
